// Generator_39573828665631
// MI455X (gfx1250) — matrix operand order repaired and checked by reading, not compiled, not run on hardware
//
#include <hip/hip_runtime.h>
#include <cmath>

typedef __attribute__((ext_vector_type(16))) _Float16 v16h;
typedef __attribute__((ext_vector_type(8)))  float    v8f;

#define SQRT2F 1.41421356237309515f
#define OT 8   // output-channel tiles (of 16) per wave; all Couts divisible by 128

__device__ __forceinline__ float lrelu_s2(float v) {
  return (v > 0.f ? v : 0.2f * v) * SQRT2F;
}

// ---------------------------------------------------------------------------
// Mapping network: PixelNorm + 8 x EqualLinear(fused lrelu, lr_mul=0.01)
// ---------------------------------------------------------------------------
__global__ __launch_bounds__(512) void gen_mapping(
    const float* __restrict__ z,
    const float* w0, const float* w1, const float* w2, const float* w3,
    const float* w4, const float* w5, const float* w6, const float* w7,
    const float* b0, const float* b1, const float* b2, const float* b3,
    const float* b4, const float* b5, const float* b6, const float* b7,
    float* __restrict__ wlat)
{
  __shared__ float h[4][512];
  __shared__ float nrm[4];
  int t = threadIdx.x;
  for (int b = 0; b < 4; ++b) h[b][t] = z[b * 512 + t];
  __syncthreads();
  if (t < 4) {
    float ss = 0.f;
    for (int i = 0; i < 512; ++i) ss += h[t][i] * h[t][i];
    nrm[t] = rsqrtf(ss * (1.f / 512.f) + 1e-8f);
  }
  __syncthreads();
  for (int b = 0; b < 4; ++b) h[b][t] *= nrm[b];
  __syncthreads();
  const float* Ws[8] = {w0, w1, w2, w3, w4, w5, w6, w7};
  const float* Bs[8] = {b0, b1, b2, b3, b4, b5, b6, b7};
  const float mscale = 0.01f * 0.044194173824159216f;  // 0.01 / sqrt(512)
  for (int l = 0; l < 8; ++l) {
    const float* wr = Ws[l] + (size_t)t * 512;
    float bias = Bs[l][t] * 0.01f;
    float acc[4];
    for (int b = 0; b < 4; ++b) {
      float a = 0.f;
      for (int i = 0; i < 512; ++i) a += h[b][i] * wr[i];
      acc[b] = lrelu_s2(a * mscale + bias);
    }
    __syncthreads();
    for (int b = 0; b < 4; ++b) h[b][t] = acc[b];
    __syncthreads();
  }
  for (int b = 0; b < 4; ++b) wlat[b * 512 + t] = h[b][t];
}

// ---------------------------------------------------------------------------
// Broadcast learned constant (1,512,4,4) -> (4,512,4,4)
// ---------------------------------------------------------------------------
__global__ void gen_tile_const(const float* __restrict__ c,
                               float* __restrict__ x, int n) {
  int i = blockIdx.x * blockDim.x + threadIdx.x;
  if (i < 4 * n) x[i] = c[i % n];
}

// ---------------------------------------------------------------------------
// Zero a small f16 region (zero page for padded conv taps)
// ---------------------------------------------------------------------------
__global__ void gen_zero_h(_Float16* __restrict__ p, int n) {
  int i = blockIdx.x * blockDim.x + threadIdx.x;
  if (i < n) p[i] = (_Float16)0.0f;
}

// ---------------------------------------------------------------------------
// Style: s[b,c] = dot(wlat[b], mod_w[c]) / sqrt(512) + mod_b[c]
// ---------------------------------------------------------------------------
__global__ __launch_bounds__(256) void gen_style(
    const float* __restrict__ wlat, const float* __restrict__ mw,
    const float* __restrict__ mb, float* __restrict__ sout, int Cin)
{
  int idx = blockIdx.x * 256 + threadIdx.x;
  if (idx >= 4 * Cin) return;
  int b = idx / Cin, c = idx - b * Cin;
  const float* hb = wlat + b * 512;
  const float* wr = mw + (size_t)c * 512;
  float a = 0.f;
  for (int i = 0; i < 512; ++i) a += hb[i] * wr[i];
  sout[idx] = a * 0.044194173824159216f + mb[c];
}

// ---------------------------------------------------------------------------
// Demod: rsqrt( sum_{i,ky,kx} (wscale*w[o,i,ky,kx]*s[b,i])^2 + 1e-8 )
// grid = (Cout, 4)
// ---------------------------------------------------------------------------
__global__ __launch_bounds__(256) void gen_demod(
    const float* __restrict__ w, const float* __restrict__ s,
    float* __restrict__ dout, int Cin, int Cout, float wscale)
{
  int o = blockIdx.x, b = blockIdx.y;
  __shared__ float red[256];
  int n = Cin * 9;
  const float* wr = w + (size_t)o * n;
  const float* sb = s + b * Cin;
  float a = 0.f;
  for (int i = threadIdx.x; i < n; i += 256) {
    float t = wscale * wr[i] * sb[i / 9];
    a += t * t;
  }
  red[threadIdx.x] = a;
  __syncthreads();
  for (int st = 128; st > 0; st >>= 1) {
    if (threadIdx.x < st) red[threadIdx.x] += red[threadIdx.x + st];
    __syncthreads();
  }
  if (threadIdx.x == 0) dout[b * Cout + o] = rsqrtf(red[0] + 1e-8f);
}

// ---------------------------------------------------------------------------
// Pre-modulate + f16 convert + NCHW->NHWC transpose:
//   xm[((b*npix + p)*Cin) + c] = (half)(x[(b*Cin + c)*npix + p] * s[b,c])
// NHWC makes the WMMA B-operand (16 consecutive channels per lane) one
// contiguous 32-byte load.
// ---------------------------------------------------------------------------
__global__ void gen_modx(const float* __restrict__ x,
                         const float* __restrict__ s,
                         _Float16* __restrict__ xm, int Cin, int npix)
{
  size_t total = (size_t)4 * Cin * npix;
  for (size_t i = blockIdx.x * (size_t)blockDim.x + threadIdx.x; i < total;
       i += (size_t)gridDim.x * blockDim.x) {
    int c = (int)(i % Cin);
    size_t r = i / Cin;
    int p = (int)(r % npix);
    int b = (int)(r / npix);
    float v = x[((size_t)b * Cin + c) * npix + p] * s[b * Cin + c];
    xm[i] = (_Float16)v;
  }
}

// ---------------------------------------------------------------------------
// Pack conv weights (pre-scaled, optionally spatially flipped) into f16 in the
// WMMA A-matrix (16x32 f16) lane layout used here: k = 16*(lane/16) + j.
// Layout: [otile][tap][chunk][lane(32)][j(16)]
// ---------------------------------------------------------------------------
__global__ void gen_pack_w(const float* __restrict__ w,
                           _Float16* __restrict__ wp,
                           int Cin, int Cout, float wscale, int flip)
{
  int CH = Cin / 32;
  size_t total = (size_t)(Cout / 16) * 9 * CH * 32 * 16;
  for (size_t i = blockIdx.x * (size_t)blockDim.x + threadIdx.x; i < total;
       i += (size_t)gridDim.x * blockDim.x) {
    int j = (int)(i & 15);
    int lane = (int)((i >> 4) & 31);
    size_t r = i >> 9;            // /512
    int chunk = (int)(r % CH); r /= CH;
    int tap = (int)(r % 9);
    int otile = (int)(r / 9);
    int hi = lane >> 4, lm = lane & 15;
    int v = j >> 1, p = j & 1;
    int k = hi * 16 + (v << 1) + p;
    int ci = chunk * 32 + k;
    int o = otile * 16 + lm;
    int tw = flip ? (8 - tap) : tap;
    wp[i] = (_Float16)(w[((size_t)o * Cin + ci) * 9 + tw] * wscale);
  }
}

// ---------------------------------------------------------------------------
// Regular modulated 3x3 conv (stride 1, pad 1) as implicit GEMM on WMMA.
// One wave computes OT*16(Cout) x 16(pixel).  B loads are contiguous v16h
// from NHWC activations; padded taps read a zero page (pointer select, no
// exec divergence).  Fused epilogue: *demod + noise + bias, lrelu*sqrt2.
// grid = (npix/16, Cout/(16*OT), B)
// ---------------------------------------------------------------------------
__global__ __launch_bounds__(32) void gen_conv3x3_wmma(
    const _Float16* __restrict__ xm,   // NHWC [B][npix][Cin]
    const _Float16* __restrict__ zp,   // >= Cin zeros
    const _Float16* __restrict__ wp,
    const float* __restrict__ demod, const float* __restrict__ noise,
    const float* __restrict__ nw, const float* __restrict__ ab,
    float* __restrict__ out, int Cin, int Cout, int H, int W)
{
  int lane = threadIdx.x;
  int hi = lane >> 4, lm = lane & 15;
  int npix = H * W;
  int CH = Cin >> 5;
  int pix = blockIdx.x * 16 + lm;
  int otbase = blockIdx.y * OT, b = blockIdx.z;
  int py = pix / W, px = pix - py * W;
  const _Float16* xb = xm + (size_t)b * npix * Cin;
  v8f acc[OT];
  #pragma unroll
  for (int ot = 0; ot < OT; ++ot) acc[ot] = (v8f){};

  #pragma unroll
  for (int tap = 0; tap < 9; ++tap) {
    int dy = tap / 3 - 1, dx = tap % 3 - 1;
    int iy = py + dy, ix = px + dx;
    bool inb = iy >= 0 && iy < H && ix >= 0 && ix < W;
    const _Float16* base =
        (inb ? (xb + (size_t)(iy * W + ix) * Cin) : zp) + hi * 16;
    const _Float16* wt = wp + (((size_t)otbase * 9 + tap) * CH) * 512 + lane * 16;
    for (int chunk = 0; chunk < CH; ++chunk) {
      v16h Bm = *(const v16h*)(base + chunk * 32);
      #pragma unroll
      for (int ot = 0; ot < OT; ++ot) {
        v16h A = *(const v16h*)(wt + ((size_t)ot * 9 * CH + chunk) * 512);
        acc[ot] = __builtin_amdgcn_wmma_f32_16x16x32_f16(
            false, A, false, Bm, (short)0, acc[ot], false, false);
      }
    }
  }
  float nv = nw[0] * noise[(size_t)b * npix + pix];
  #pragma unroll
  for (int ot = 0; ot < OT; ++ot) {
    #pragma unroll
    for (int r = 0; r < 8; ++r) {
      int m_ = r + 8 * hi;   // C/D: lanes 0-15 M=r, lanes 16-31 M=8+r
      int oo = (otbase + ot) * 16 + m_;
      float v = acc[ot][r] * demod[b * Cout + oo] + nv + ab[oo];
      out[((size_t)b * Cout + oo) * npix + pix] = lrelu_s2(v);
    }
  }
}

// ---------------------------------------------------------------------------
// Transposed (stride-2) modulated conv -> out_t of (2H+1)x(2W+1); weights
// pre-flipped by the pack kernel.  Raw f32 out (demod/blur/epilogue follow).
// grid = (ceil(npixt/16), Cout/(16*OT), B)
// ---------------------------------------------------------------------------
__global__ __launch_bounds__(32) void gen_convup_wmma(
    const _Float16* __restrict__ xm,   // NHWC [B][H*W][Cin]
    const _Float16* __restrict__ zp,
    const _Float16* __restrict__ wp,
    float* __restrict__ outt, int Cin, int Cout, int H, int W)
{
  int lane = threadIdx.x;
  int hi = lane >> 4, lm = lane & 15;
  int Ht = 2 * H + 1, Wt = 2 * W + 1;
  int npixt = Ht * Wt;
  int CH = Cin >> 5;
  int pix = blockIdx.x * 16 + lm;
  int otbase = blockIdx.y * OT, b = blockIdx.z;
  int py = pix / Wt, px = pix - py * Wt;
  const _Float16* xb = xm + (size_t)b * H * W * Cin;
  v8f acc[OT];
  #pragma unroll
  for (int ot = 0; ot < OT; ++ot) acc[ot] = (v8f){};

  #pragma unroll
  for (int tap = 0; tap < 9; ++tap) {
    int ky = tap / 3, kx = tap - ky * 3;
    int ty = py + ky - 2, tx = px + kx - 2;
    bool inb = (pix < npixt) && ty >= 0 && tx >= 0 && !(ty & 1) && !(tx & 1);
    int iy = ty >> 1, ix = tx >> 1;
    inb = inb && iy < H && ix < W;
    int iyc = min(max(iy, 0), H - 1);
    int ixc = min(max(ix, 0), W - 1);
    const _Float16* base =
        (inb ? (xb + (size_t)(iyc * W + ixc) * Cin) : zp) + hi * 16;
    const _Float16* wt = wp + (((size_t)otbase * 9 + tap) * CH) * 512 + lane * 16;
    for (int chunk = 0; chunk < CH; ++chunk) {
      v16h Bm = *(const v16h*)(base + chunk * 32);
      #pragma unroll
      for (int ot = 0; ot < OT; ++ot) {
        v16h A = *(const v16h*)(wt + ((size_t)ot * 9 * CH + chunk) * 512);
        acc[ot] = __builtin_amdgcn_wmma_f32_16x16x32_f16(
            false, A, false, Bm, (short)0, acc[ot], false, false);
      }
    }
  }
  if (pix < npixt) {
    #pragma unroll
    for (int ot = 0; ot < OT; ++ot) {
      #pragma unroll
      for (int r = 0; r < 8; ++r) {
        int m_ = r + 8 * hi;
        int oo = (otbase + ot) * 16 + m_;
        outt[((size_t)b * Cout + oo) * npixt + pix] = acc[ot][r];
      }
    }
  }
}

// ---------------------------------------------------------------------------
// Blur(upsample_factor=2): 4x4 FIR (kernel*4, pad(1,1)) over out_t, fused with
// demod, noise injection, bias and leaky_relu*sqrt(2).
// ---------------------------------------------------------------------------
__global__ void gen_blur_epi(const float* __restrict__ t,
                             const float* __restrict__ demod,
                             const float* __restrict__ noise,
                             const float* __restrict__ nw,
                             const float* __restrict__ ab,
                             float* __restrict__ out,
                             int Cout, int H2, int Ht)
{
  const float c4[4] = {0.25f, 0.75f, 0.75f, 0.25f};
  size_t total = (size_t)4 * Cout * H2 * H2;
  for (size_t i = blockIdx.x * (size_t)blockDim.x + threadIdx.x; i < total;
       i += (size_t)gridDim.x * blockDim.x) {
    int xx = (int)(i % H2);
    size_t r = i / H2;
    int yy = (int)(r % H2); r /= H2;
    int c = (int)(r % Cout);
    int b = (int)(r / Cout);
    const float* tb = t + ((size_t)b * Cout + c) * Ht * Ht;
    float a = 0.f;
    for (int fy = 0; fy < 4; ++fy) {
      int iy = yy + fy - 1;
      if (iy < 0 || iy >= Ht) continue;
      for (int fx = 0; fx < 4; ++fx) {
        int ix = xx + fx - 1;
        if (ix < 0 || ix >= Ht) continue;
        a += c4[fy] * c4[fx] * tb[(size_t)iy * Ht + ix];
      }
    }
    float v = a * demod[b * Cout + c] +
              nw[0] * noise[(size_t)b * H2 * H2 + (size_t)yy * H2 + xx] + ab[c];
    out[i] = lrelu_s2(v);
  }
}

// ---------------------------------------------------------------------------
// Skip upsample: upfirdn2d(up=2, pad=(2,1)), kernel*4.  (B,3,H,H)->(B,3,2H,2H)
// ---------------------------------------------------------------------------
__global__ void gen_fir_up2(const float* __restrict__ skip,
                            float* __restrict__ out, int H, int C)
{
  const float c4[4] = {0.25f, 0.75f, 0.75f, 0.25f};
  int H2 = 2 * H;
  size_t total = (size_t)4 * C * H2 * H2;
  for (size_t i = blockIdx.x * (size_t)blockDim.x + threadIdx.x; i < total;
       i += (size_t)gridDim.x * blockDim.x) {
    int xx = (int)(i % H2);
    size_t r = i / H2;
    int yy = (int)(r % H2); r /= H2;
    int c = (int)(r % C);
    int b = (int)(r / C);
    const float* sb = skip + ((size_t)b * C + c) * H * H;
    float a = 0.f;
    for (int fy = 0; fy < 4; ++fy) {
      int ty = yy + fy - 2;
      if (ty < 0 || (ty & 1)) continue;
      int iy = ty >> 1;
      if (iy >= H) continue;
      for (int fx = 0; fx < 4; ++fx) {
        int tx = xx + fx - 2;
        if (tx < 0 || (tx & 1)) continue;
        int ix = tx >> 1;
        if (ix >= H) continue;
        a += c4[fy] * c4[fx] * sb[(size_t)iy * H + ix];
      }
    }
    out[i] = a;
  }
}

// ---------------------------------------------------------------------------
// toRGB: 1x1 modulated conv, NO demod, + upsampled skip
// ---------------------------------------------------------------------------
__global__ void gen_torgb(const float* __restrict__ x,
                          const float* __restrict__ s,
                          const float* __restrict__ rw,
                          const float* __restrict__ rb,
                          const float* __restrict__ skipup,
                          float* __restrict__ out,
                          int Cin, int H, float rscale)
{
  size_t npix = (size_t)H * H;
  size_t total = 4 * 3 * npix;
  for (size_t i = blockIdx.x * (size_t)blockDim.x + threadIdx.x; i < total;
       i += (size_t)gridDim.x * blockDim.x) {
    size_t p = i % npix;
    size_t r = i / npix;
    int o = (int)(r % 3);
    int b = (int)(r / 3);
    const float* xb = x + (size_t)b * Cin * npix + p;
    const float* sb = s + b * Cin;
    const float* wr = rw + (size_t)o * Cin;
    float a = 0.f;
    for (int ci = 0; ci < Cin; ++ci) a += xb[(size_t)ci * npix] * sb[ci] * wr[ci];
    a = a * rscale + rb[o];
    if (skipup) a += skipup[i];
    out[i] = a;
  }
}

// ---------------------------------------------------------------------------
// Host orchestration
// ---------------------------------------------------------------------------
extern "C" void kernel_launch(void* const* d_in, const int* in_sizes, int n_in,
                              void* d_out, int out_size, void* d_ws, size_t ws_size,
                              hipStream_t stream) {
  (void)in_sizes; (void)n_in; (void)out_size; (void)ws_size;
  const float* z = (const float*)d_in[0];
  const float* mlp_w[8]; const float* mlp_b[8];
  for (int i = 0; i < 8; ++i) {
    mlp_w[i] = (const float*)d_in[1 + i];
    mlp_b[i] = (const float*)d_in[9 + i];
  }
  const float* cst = (const float*)d_in[17];
  const float* conv_w[13]; const float* mod_w[13]; const float* mod_b[13];
  const float* noise_w[13]; const float* act_b[13]; const float* noises[13];
  for (int i = 0; i < 13; ++i) {
    conv_w[i]  = (const float*)d_in[18 + i];
    mod_w[i]   = (const float*)d_in[31 + i];
    mod_b[i]   = (const float*)d_in[44 + i];
    noise_w[i] = (const float*)d_in[57 + i];
    act_b[i]   = (const float*)d_in[70 + i];
    noises[i]  = (const float*)d_in[111 + i];
  }
  const float* rgb_w[7]; const float* rgb_mw[7]; const float* rgb_mb[7]; const float* rgb_b[7];
  for (int i = 0; i < 7; ++i) {
    rgb_w[i]  = (const float*)d_in[83 + i];
    rgb_mw[i] = (const float*)d_in[90 + i];
    rgb_mb[i] = (const float*)d_in[97 + i];
    rgb_b[i]  = (const float*)d_in[104 + i];
  }

  float* wsf = (float*)d_ws;
  size_t off = 0;
  auto alloc = [&](size_t n) { float* p = wsf + off; off += n; return p; };
  float* wlat     = alloc(4 * 512);
  float* styleBuf = alloc(4 * 512);
  float* demodBuf = alloc(4 * 512);
  const size_t ACT  = (size_t)4 * 128 * 256 * 256;   // max activation (f32)
  const size_t ACTT = (size_t)4 * 128 * 257 * 257;   // max conv-transpose out (f32)
  float* bufA = alloc(ACT);
  float* bufB = alloc(ACT);
  float* bufT = alloc(ACTT);
  const size_t SK = (size_t)4 * 3 * 256 * 256;
  float* skipA = alloc(SK);
  float* skipB = alloc(SK);
  float* skipU = alloc(SK);
  _Float16* xmh   = (_Float16*)alloc(ACT / 2);                    // f16 NHWC acts
  _Float16* zpage = (_Float16*)alloc(512);                        // 1024 zero halfs
  _Float16* wpack = (_Float16*)alloc((size_t)9 * 512 * 512 / 2);  // f16 packed W

  static const int cin_[13]  = {512, 512, 512, 512, 512, 512, 512, 512, 512, 512, 256, 256, 128};
  static const int cout_[13] = {512, 512, 512, 512, 512, 512, 512, 512, 512, 256, 256, 128, 128};

  gen_zero_h<<<4, 256, 0, stream>>>(zpage, 1024);

  gen_mapping<<<1, 512, 0, stream>>>(z,
      mlp_w[0], mlp_w[1], mlp_w[2], mlp_w[3], mlp_w[4], mlp_w[5], mlp_w[6], mlp_w[7],
      mlp_b[0], mlp_b[1], mlp_b[2], mlp_b[3], mlp_b[4], mlp_b[5], mlp_b[6], mlp_b[7],
      wlat);

  { int n = 512 * 16; gen_tile_const<<<(4 * n + 255) / 256, 256, 0, stream>>>(cst, bufA, n); }

  float* x = bufA;
  float* nxt = bufB;
  int H = 4;

  auto launch_style = [&](const float* mw, const float* mb, int Cin) {
    gen_style<<<(4 * Cin + 255) / 256, 256, 0, stream>>>(wlat, mw, mb, styleBuf, Cin);
  };
  auto prep_layer = [&](int ci, int flip) {  // style -> modx -> pack -> demod
    int Cin = cin_[ci], Cout = cout_[ci];
    float wsc = 1.0f / sqrtf((float)(Cin * 9));
    launch_style(mod_w[ci], mod_b[ci], Cin);
    size_t nx = (size_t)4 * Cin * H * H;
    gen_modx<<<(int)((nx + 255) / 256), 256, 0, stream>>>(x, styleBuf, xmh, Cin, H * H);
    size_t np = (size_t)9 * Cout * Cin;
    gen_pack_w<<<(int)((np + 255) / 256), 256, 0, stream>>>(conv_w[ci], wpack, Cin, Cout, wsc, flip);
    gen_demod<<<dim3(Cout, 4), 256, 0, stream>>>(conv_w[ci], styleBuf, demodBuf, Cin, Cout, wsc);
  };
  auto launch_conv = [&](int ci) {  // regular 3x3 conv, swaps x/nxt
    int Cin = cin_[ci], Cout = cout_[ci];
    int npix = H * H;
    gen_conv3x3_wmma<<<dim3(npix / 16, Cout / (16 * OT), 4), 32, 0, stream>>>(
        xmh, zpage, wpack, demodBuf, noises[ci], noise_w[ci], act_b[ci],
        nxt, Cin, Cout, H, H);
    float* t = x; x = nxt; nxt = t;
  };

  // conv 0 (4x4, no up) + rgb 0
  prep_layer(0, 0);
  launch_conv(0);
  launch_style(rgb_mw[0], rgb_mb[0], 512);
  {
    size_t tot = (size_t)4 * 3 * H * H;
    gen_torgb<<<(int)((tot + 255) / 256), 256, 0, stream>>>(
        x, styleBuf, rgb_w[0], rgb_b[0], nullptr, skipA, 512, H, 1.0f / sqrtf(512.f));
  }
  float* skip = skipA;
  float* skipNext = skipB;

  for (int j = 1; j <= 6; ++j) {
    int ci = 2 * j - 1;          // upsample conv
    int Cin = cin_[ci], Cout = cout_[ci];
    prep_layer(ci, 1);           // pre-flip weights for transposed conv
    {
      int Ht = 2 * H + 1;
      int npixt = Ht * Ht;
      gen_convup_wmma<<<dim3((npixt + 15) / 16, Cout / (16 * OT), 4), 32, 0, stream>>>(
          xmh, zpage, wpack, bufT, Cin, Cout, H, H);
      int H2 = 2 * H;
      size_t tot = (size_t)4 * Cout * H2 * H2;
      gen_blur_epi<<<(int)((tot + 255) / 256), 256, 0, stream>>>(
          bufT, demodBuf, noises[ci], noise_w[ci], act_b[ci], nxt, Cout, H2, Ht);
      float* t = x; x = nxt; nxt = t;
      H = H2;
    }
    int ci2 = ci + 1;            // regular conv
    prep_layer(ci2, 0);
    launch_conv(ci2);
    {                            // toRGB with upsampled skip
      int Hs = H / 2;
      size_t tot = (size_t)4 * 3 * H * H;
      gen_fir_up2<<<(int)((tot + 255) / 256), 256, 0, stream>>>(skip, skipU, Hs, 3);
      float* outp = (j == 6) ? (float*)d_out : skipNext;
      int Crgb = cout_[ci2];
      launch_style(rgb_mw[j], rgb_mb[j], Crgb);
      gen_torgb<<<(int)((tot + 255) / 256), 256, 0, stream>>>(
          x, styleBuf, rgb_w[j], rgb_b[j], skipU, outp, Crgb, H, 1.0f / sqrtf((float)Crgb));
      float* t2 = skip; skip = skipNext; skipNext = t2;
    }
  }
}
